// TransformerEncoderLayer_23502061043782
// MI455X (gfx1250) — compile-verified, matrix operand order checked by reading, not run on hardware
//
#include <hip/hip_runtime.h>
#include <hip/hip_bf16.h>

typedef __attribute__((ext_vector_type(16))) __bf16 v16bf;
typedef __attribute__((ext_vector_type(8)))  __bf16 v8bf;
typedef __attribute__((ext_vector_type(8)))  float  v8f;

#define DEVINL __device__ __forceinline__

DEVINL v16bf frag_cat(v8bf lo, v8bf hi) {
  return __builtin_shufflevector(lo, hi, 0,1,2,3,4,5,6,7,8,9,10,11,12,13,14,15);
}

DEVINL v8f wmma_bf16(v16bf a, v16bf b, v8f c) {
  // D = A(16x32) * B(32x16) + C(16x16), f32 accum
  return __builtin_amdgcn_wmma_f32_16x16x32_bf16(false, a, false, b, (short)0, c, false, false);
}

DEVINL v8f zero8() { v8f z = {0.f,0.f,0.f,0.f,0.f,0.f,0.f,0.f}; return z; }

// ---------------------------------------------------------------------------
// CDNA5 async memory->LDS copy (16B per lane), tracked by ASYNCcnt.
// LDS flat addresses carry the wave-relative LDS byte offset in bits [31:0].
// ---------------------------------------------------------------------------
DEVINL void async_cp_b128(void* ldsptr, const void* gptr) {
  unsigned loff = (unsigned)(uintptr_t)ldsptr;
  asm volatile("global_load_async_to_lds_b128 %0, %1, off"
               :: "v"(loff), "v"(gptr) : "memory");
}

DEVINL void wait_async0() {
#if __has_builtin(__builtin_amdgcn_s_wait_asynccnt)
  __builtin_amdgcn_s_wait_asynccnt(0);
#else
  asm volatile("s_wait_asynccnt 0" ::: "memory");
#endif
}

// ---------------------------------------------------------------------------
// f32 -> bf16 convert
// ---------------------------------------------------------------------------
__global__ __launch_bounds__(256) void cvt_bf16_kernel(const float* __restrict__ in,
                                                       __bf16* __restrict__ out, int n) {
  int i = blockIdx.x * 256 + threadIdx.x;
  if (i < n) out[i] = (__bf16)in[i];
}

// ---------------------------------------------------------------------------
// LayerNorm over last dim (1024), fp32 in -> bf16 out
// ---------------------------------------------------------------------------
__global__ __launch_bounds__(256) void ln_bf16_kernel(const float* __restrict__ x,
                                                      const float* __restrict__ g,
                                                      const float* __restrict__ b,
                                                      __bf16* __restrict__ out) {
  const int row = blockIdx.x;
  const int tid = threadIdx.x;
  const float* xr = x + (size_t)row * 1024;
  float xv[4];
  float s1 = 0.f, s2 = 0.f;
#pragma unroll
  for (int i = 0; i < 4; i++) {
    xv[i] = xr[tid + i * 256];
    s1 += xv[i];
    s2 += xv[i] * xv[i];
  }
#pragma unroll
  for (int off = 16; off > 0; off >>= 1) {
    s1 += __shfl_down(s1, off);
    s2 += __shfl_down(s2, off);
  }
  __shared__ float red[16];
  const int lane = tid & 31, wv = tid >> 5;
  if (lane == 0) { red[wv] = s1; red[8 + wv] = s2; }
  __syncthreads();
  if (tid == 0) {
    float a = 0.f, q = 0.f;
    for (int i = 0; i < 8; i++) { a += red[i]; q += red[8 + i]; }
    red[0] = a; red[8] = q;
  }
  __syncthreads();
  const float mean = red[0] * (1.f / 1024.f);
  const float var  = red[8] * (1.f / 1024.f) - mean * mean;
  const float rstd = rsqrtf(var + 1e-5f);
  __bf16* orow = out + (size_t)row * 1024;
#pragma unroll
  for (int i = 0; i < 4; i++) {
    int c = tid + i * 256;
    orow[c] = (__bf16)((xv[i] - mean) * rstd * g[c] + b[c]);
  }
}

// ---------------------------------------------------------------------------
// Generic WMMA GEMM: C[M,N] = A[M,K] @ B[N,K]^T  (A,B bf16 row-major)
// 128x128 tile / block, 8 waves, each wave 64x32 (4x2 wmma frags).
// Double-buffered LDS fed by async global->LDS copies (ASYNCcnt).
// ---------------------------------------------------------------------------
enum { EPI_PHI = 0, EPI_SCALE = 1, EPI_GELU = 2, EPI_RES = 3 };

template <int EPI>
__global__ __launch_bounds__(256) void gemm_bf16_kernel(
    const __bf16* __restrict__ A, const __bf16* __restrict__ B,
    int K, int lda, int ldb, int ldc,
    const float* __restrict__ bias, const float* __restrict__ resid,
    __bf16* __restrict__ outb, float* __restrict__ outf, float scale) {
  __shared__ __bf16 As[2][128][40];
  __shared__ __bf16 Bs[2][128][40];
  const int tid = threadIdx.x;
  const int lane = tid & 31, wave = tid >> 5;
  const int wm = wave >> 2, wn = wave & 3;   // 2 x 4 wave grid
  const int m0 = blockIdx.y * 128, n0 = blockIdx.x * 128;
  const int c0 = (lane >> 4) << 3;           // K-chunk base: 0 or 8
  const int mrow = lane & 15;

  v8f acc[4][2];
#pragma unroll
  for (int r = 0; r < 4; r++)
#pragma unroll
    for (int c = 0; c < 2; c++) acc[r][c] = zero8();

  // staging geometry: 512 chunks of 8 bf16 per matrix, 2 per thread
  const int ch_row0 = tid >> 2;
  const int ch_col0 = (tid & 3) << 3;
  const int ch_row1 = (tid + 256) >> 2;
  const int ch_col1 = ((tid + 256) & 3) << 3;

  auto stage = [&](int buf, int k0) {
    async_cp_b128(&As[buf][ch_row0][ch_col0],
                  &A[(size_t)(m0 + ch_row0) * lda + k0 + ch_col0]);
    async_cp_b128(&Bs[buf][ch_row0][ch_col0],
                  &B[(size_t)(n0 + ch_row0) * ldb + k0 + ch_col0]);
    async_cp_b128(&As[buf][ch_row1][ch_col1],
                  &A[(size_t)(m0 + ch_row1) * lda + k0 + ch_col1]);
    async_cp_b128(&Bs[buf][ch_row1][ch_col1],
                  &B[(size_t)(n0 + ch_row1) * ldb + k0 + ch_col1]);
  };

  stage(0, 0);
  wait_async0();
  __syncthreads();

  int buf = 0;
  for (int k0 = 0; k0 < K; k0 += 32) {
    if (k0 + 32 < K) stage(buf ^ 1, k0 + 32);  // overlap next tile with compute

    v16bf af[4], bfm[2];
#pragma unroll
    for (int r = 0; r < 4; r++) {
      const __bf16* p = &As[buf][wm * 64 + r * 16 + mrow][0];
      af[r] = frag_cat(*(const v8bf*)(p + c0), *(const v8bf*)(p + c0 + 16));
    }
#pragma unroll
    for (int c = 0; c < 2; c++) {
      const __bf16* p = &Bs[buf][wn * 32 + c * 16 + mrow][0];
      bfm[c] = frag_cat(*(const v8bf*)(p + c0), *(const v8bf*)(p + c0 + 16));
    }
#pragma unroll
    for (int r = 0; r < 4; r++)
#pragma unroll
      for (int c = 0; c < 2; c++) acc[r][c] = wmma_bf16(af[r], bfm[c], acc[r][c]);

    wait_async0();
    __syncthreads();
    buf ^= 1;
  }

#pragma unroll
  for (int r = 0; r < 4; r++)
#pragma unroll
    for (int c = 0; c < 2; c++)
#pragma unroll
      for (int v = 0; v < 8; v++) {
        int mm = m0 + wm * 64 + r * 16 + ((lane >> 4) << 3) + v;
        int nn = n0 + wn * 32 + c * 16 + (lane & 15);
        float x = acc[r][c][v];
        size_t off = (size_t)mm * ldc + nn;
        if constexpr (EPI == EPI_PHI) {
          // elu(x)+1
          outb[off] = (__bf16)(x > 0.f ? x + 1.f : __expf(x));
        } else if constexpr (EPI == EPI_SCALE) {
          outb[off] = (__bf16)(x * scale);
        } else if constexpr (EPI == EPI_GELU) {
          x += bias[nn];
          outb[off] = (__bf16)(0.5f * x * (1.f + erff(x * 0.70710678118f)));
        } else {
          outf[off] = x + bias[nn] + resid[off];
        }
      }
}

// ---------------------------------------------------------------------------
// Ksum[n,c] = sum_l phiK[n,l,c]
// ---------------------------------------------------------------------------
__global__ __launch_bounds__(256) void ksum_kernel(const __bf16* __restrict__ kphi,
                                                   float* __restrict__ ks) {
  const int c = blockIdx.x * 256 + threadIdx.x;
  const int n = blockIdx.y;
  const __bf16* p = kphi + (size_t)n * 4096 * 1024 + c;
  float s = 0.f;
  for (int l = 0; l < 4096; l++) s += (float)p[(size_t)l * 1024];
  ks[n * 1024 + c] = s;
}

// ---------------------------------------------------------------------------
// Z[n,l,h] = 1/(phiQ[n,l,h,:] . Ksum[n,h,:] + eps)
// ---------------------------------------------------------------------------
__global__ __launch_bounds__(256) void z_kernel(const __bf16* __restrict__ qphi,
                                                const float* __restrict__ ks,
                                                float* __restrict__ z) {
  const int i = blockIdx.x * 256 + threadIdx.x;  // [0, 2*4096*8)
  const int h = i & 7;
  const int l = (i >> 3) & 4095;
  const int n = i >> 15;
  const __bf16* q = qphi + ((size_t)(n * 4096 + l)) * 1024 + h * 128;
  const float* K = ks + n * 1024 + h * 128;
  float s = 0.f;
#pragma unroll 8
  for (int d = 0; d < 128; d++) s += (float)q[d] * K[d];
  z[i] = 1.f / (s + 1e-6f);
}

// ---------------------------------------------------------------------------
// KV^T[n,h,e,d] = sum_s V[n,s,h,e] * K[n,s,h,d]   (= (K^T V)^T), bf16 out
// One block per (n,h); both operands transpose-staged through LDS.
// ---------------------------------------------------------------------------
__global__ __launch_bounds__(256) void kv_kernel(const __bf16* __restrict__ kphi,
                                                 const __bf16* __restrict__ vs,
                                                 __bf16* __restrict__ kvt) {
  const int h = blockIdx.x, n = blockIdx.y;
  __shared__ __bf16 Vt[128][40];  // [e][s]
  __shared__ __bf16 Kt[128][40];  // [d][s]
  const int tid = threadIdx.x;
  const int lane = tid & 31, wave = tid >> 5;
  const int wm = wave >> 2, wn = wave & 3;
  const int c0 = (lane >> 4) << 3;
  const int mrow = lane & 15;
  const size_t base = (size_t)n * 4096 * 1024 + (size_t)h * 128;

  v8f acc[4][2];
#pragma unroll
  for (int r = 0; r < 4; r++)
#pragma unroll
    for (int c = 0; c < 2; c++) acc[r][c] = zero8();

  for (int s0 = 0; s0 < 4096; s0 += 32) {
#pragma unroll
    for (int i = 0; i < 2; i++) {
      int ch = tid + i * 256;        // 512 chunks: 32 rows x 16 col-groups
      int sl = ch >> 4;              // 0..31  (s within tile)
      int col = (ch & 15) << 3;      // 0..120 (e/d within head)
      v8bf vv = *(const v8bf*)&vs[base + (size_t)(s0 + sl) * 1024 + col];
      v8bf kk = *(const v8bf*)&kphi[base + (size_t)(s0 + sl) * 1024 + col];
#pragma unroll
      for (int j = 0; j < 8; j++) {
        Vt[col + j][sl] = vv[j];
        Kt[col + j][sl] = kk[j];
      }
    }
    __syncthreads();
    v16bf af[4], bfm[2];
#pragma unroll
    for (int r = 0; r < 4; r++) {
      const __bf16* p = &Vt[wm * 64 + r * 16 + mrow][0];
      af[r] = frag_cat(*(const v8bf*)(p + c0), *(const v8bf*)(p + c0 + 16));
    }
#pragma unroll
    for (int c = 0; c < 2; c++) {
      const __bf16* p = &Kt[wn * 32 + c * 16 + mrow][0];
      bfm[c] = frag_cat(*(const v8bf*)(p + c0), *(const v8bf*)(p + c0 + 16));
    }
#pragma unroll
    for (int r = 0; r < 4; r++)
#pragma unroll
      for (int c = 0; c < 2; c++) acc[r][c] = wmma_bf16(af[r], bfm[c], acc[r][c]);
    __syncthreads();
  }

  __bf16* out = kvt + ((size_t)(n * 8 + h)) * 128 * 128;
#pragma unroll
  for (int r = 0; r < 4; r++)
#pragma unroll
    for (int c = 0; c < 2; c++)
#pragma unroll
      for (int v = 0; v < 8; v++) {
        int mm = wm * 64 + r * 16 + ((lane >> 4) << 3) + v;  // e
        int nn = wn * 32 + c * 16 + (lane & 15);             // d
        out[mm * 128 + nn] = (__bf16)acc[r][c][v];
      }
}

// ---------------------------------------------------------------------------
// out = Q @ KV with Z*L scaling + residual:  xmid = x + (phiQ @ KV^T') * z * L
// blocks: (h, m-tile, n). B = KV^T layout [e][d] is load-direct.
// Double-buffered async staging.
// ---------------------------------------------------------------------------
__global__ __launch_bounds__(256) void attn_out_kernel(
    const __bf16* __restrict__ qphi, const __bf16* __restrict__ kvt,
    const float* __restrict__ z, const float* __restrict__ x,
    float* __restrict__ xmid) {
  const int h = blockIdx.x, mt = blockIdx.y, n = blockIdx.z;
  const int m0 = mt * 128;
  __shared__ __bf16 As[2][128][40];
  __shared__ __bf16 Bs[2][128][40];
  const int tid = threadIdx.x;
  const int lane = tid & 31, wave = tid >> 5;
  const int wm = wave >> 2, wn = wave & 3;
  const int c0 = (lane >> 4) << 3;
  const int mrow = lane & 15;
  const size_t abase = (size_t)n * 4096 * 1024 + (size_t)h * 128;
  const __bf16* Bmat = kvt + ((size_t)(n * 8 + h)) * 128 * 128;

  v8f acc[4][2];
#pragma unroll
  for (int r = 0; r < 4; r++)
#pragma unroll
    for (int c = 0; c < 2; c++) acc[r][c] = zero8();

  const int ch_row0 = tid >> 2;
  const int ch_col0 = (tid & 3) << 3;
  const int ch_row1 = (tid + 256) >> 2;
  const int ch_col1 = ((tid + 256) & 3) << 3;

  auto stage = [&](int buf, int k0) {
    async_cp_b128(&As[buf][ch_row0][ch_col0],
                  &qphi[abase + (size_t)(m0 + ch_row0) * 1024 + k0 + ch_col0]);
    async_cp_b128(&Bs[buf][ch_row0][ch_col0],
                  &Bmat[(size_t)ch_row0 * 128 + k0 + ch_col0]);
    async_cp_b128(&As[buf][ch_row1][ch_col1],
                  &qphi[abase + (size_t)(m0 + ch_row1) * 1024 + k0 + ch_col1]);
    async_cp_b128(&Bs[buf][ch_row1][ch_col1],
                  &Bmat[(size_t)ch_row1 * 128 + k0 + ch_col1]);
  };

  stage(0, 0);
  wait_async0();
  __syncthreads();

  int buf = 0;
  for (int k0 = 0; k0 < 128; k0 += 32) {
    if (k0 + 32 < 128) stage(buf ^ 1, k0 + 32);

    v16bf af[4], bfm[2];
#pragma unroll
    for (int r = 0; r < 4; r++) {
      const __bf16* p = &As[buf][wm * 64 + r * 16 + mrow][0];
      af[r] = frag_cat(*(const v8bf*)(p + c0), *(const v8bf*)(p + c0 + 16));
    }
#pragma unroll
    for (int c = 0; c < 2; c++) {
      const __bf16* p = &Bs[buf][wn * 32 + c * 16 + mrow][0];
      bfm[c] = frag_cat(*(const v8bf*)(p + c0), *(const v8bf*)(p + c0 + 16));
    }
#pragma unroll
    for (int r = 0; r < 4; r++)
#pragma unroll
      for (int c = 0; c < 2; c++) acc[r][c] = wmma_bf16(af[r], bfm[c], acc[r][c]);

    wait_async0();
    __syncthreads();
    buf ^= 1;
  }

#pragma unroll
  for (int r = 0; r < 4; r++)
#pragma unroll
    for (int c = 0; c < 2; c++)
#pragma unroll
      for (int v = 0; v < 8; v++) {
        int mm = m0 + wm * 64 + r * 16 + ((lane >> 4) << 3) + v;  // l
        int nn = wn * 32 + c * 16 + (lane & 15);                  // e
        float zz = z[((size_t)n * 4096 + mm) * 8 + h];
        size_t off = (size_t)(n * 4096 + mm) * 1024 + (size_t)h * 128 + nn;
        xmid[off] = x[off] + acc[r][c][v] * zz * 4096.f;
      }
}

// ---------------------------------------------------------------------------
// Host launcher
// ---------------------------------------------------------------------------
extern "C" void kernel_launch(void* const* d_in, const int* in_sizes, int n_in,
                              void* d_out, int out_size, void* d_ws, size_t ws_size,
                              hipStream_t stream) {
  (void)in_sizes; (void)n_in; (void)out_size; (void)ws_size;
  const float* x   = (const float*)d_in[0];
  const float* wq  = (const float*)d_in[1];
  const float* wk  = (const float*)d_in[2];
  const float* wv  = (const float*)d_in[3];
  const float* w1  = (const float*)d_in[4];
  const float* b1  = (const float*)d_in[5];
  const float* w2  = (const float*)d_in[6];
  const float* b2  = (const float*)d_in[7];
  const float* g1  = (const float*)d_in[8];
  const float* be1 = (const float*)d_in[9];
  const float* g2  = (const float*)d_in[10];
  const float* be2 = (const float*)d_in[11];
  float* out = (float*)d_out;

  const int Ntok = 8192;   // 2 * 4096
  const int C = 1024, FF = 4096;

  char* w = (char*)d_ws;
  auto alloc = [&](size_t bytes) -> char* {
    char* p = w;
    w += (bytes + 255) & ~(size_t)255;
    return p;
  };

  __bf16* wqb  = (__bf16*)alloc((size_t)C * C * 2);
  __bf16* wkb  = (__bf16*)alloc((size_t)C * C * 2);
  __bf16* wvb  = (__bf16*)alloc((size_t)C * C * 2);
  __bf16* w1b  = (__bf16*)alloc((size_t)FF * C * 2);
  __bf16* w2b  = (__bf16*)alloc((size_t)C * FF * 2);
  __bf16* x2b  = (__bf16*)alloc((size_t)Ntok * C * 2);
  __bf16* qphi = (__bf16*)alloc((size_t)Ntok * C * 2);
  __bf16* kphi = (__bf16*)alloc((size_t)Ntok * C * 2);
  __bf16* vsb  = (__bf16*)alloc((size_t)Ntok * C * 2);
  float*  ksum = (float*)alloc((size_t)2 * C * 4);
  float*  zf   = (float*)alloc((size_t)2 * 4096 * 8 * 4);
  __bf16* kvtb = (__bf16*)alloc((size_t)16 * 128 * 128 * 2);
  float*  xmid = (float*)alloc((size_t)Ntok * C * 4);
  __bf16* hb   = (__bf16*)alloc((size_t)Ntok * FF * 2);

  // weights -> bf16
  cvt_bf16_kernel<<<dim3((C * C) / 256), 256, 0, stream>>>(wq, wqb, C * C);
  cvt_bf16_kernel<<<dim3((C * C) / 256), 256, 0, stream>>>(wk, wkb, C * C);
  cvt_bf16_kernel<<<dim3((C * C) / 256), 256, 0, stream>>>(wv, wvb, C * C);
  cvt_bf16_kernel<<<dim3((FF * C) / 256), 256, 0, stream>>>(w1, w1b, FF * C);
  cvt_bf16_kernel<<<dim3((C * FF) / 256), 256, 0, stream>>>(w2, w2b, C * FF);

  // LN1
  ln_bf16_kernel<<<dim3(Ntok), 256, 0, stream>>>(x, g1, be1, x2b);

  // Q/K/V projections with fused feature maps
  gemm_bf16_kernel<EPI_PHI><<<dim3(C / 128, Ntok / 128), 256, 0, stream>>>(
      x2b, wqb, C, C, C, C, nullptr, nullptr, qphi, nullptr, 0.f);
  gemm_bf16_kernel<EPI_PHI><<<dim3(C / 128, Ntok / 128), 256, 0, stream>>>(
      x2b, wkb, C, C, C, C, nullptr, nullptr, kphi, nullptr, 0.f);
  gemm_bf16_kernel<EPI_SCALE><<<dim3(C / 128, Ntok / 128), 256, 0, stream>>>(
      x2b, wvb, C, C, C, C, nullptr, nullptr, vsb, nullptr, 1.f / 4096.f);

  // attention normalizer
  ksum_kernel<<<dim3(C / 256, 2), 256, 0, stream>>>(kphi, ksum);
  z_kernel<<<dim3((2 * 4096 * 8) / 256), 256, 0, stream>>>(qphi, ksum, zf);

  // KV^T per (n,h), then out = Q@KV with z*L + residual
  kv_kernel<<<dim3(8, 2), 256, 0, stream>>>(kphi, vsb, kvtb);
  attn_out_kernel<<<dim3(8, 4096 / 128, 2), 256, 0, stream>>>(qphi, kvtb, zf, x, xmid);

  // LN2 + MLP
  ln_bf16_kernel<<<dim3(Ntok), 256, 0, stream>>>(xmid, g2, be2, x2b);
  gemm_bf16_kernel<EPI_GELU><<<dim3(FF / 128, Ntok / 128), 256, 0, stream>>>(
      x2b, w1b, C, C, C, FF, b1, nullptr, hb, nullptr, 0.f);
  gemm_bf16_kernel<EPI_RES><<<dim3(C / 128, Ntok / 128), 256, 0, stream>>>(
      hb, w2b, FF, FF, FF, C, b2, xmid, nullptr, out, 0.f);
}
